// ScaleKD_17480516894741
// MI455X (gfx1250) — compile-verified, matrix operand order checked by reading, not run on hardware
//
#include <hip/hip_runtime.h>
#include <hip/hip_bf16.h>

// ---------------------------------------------------------------------------
// Types
// ---------------------------------------------------------------------------
typedef __bf16 bf16;
typedef __attribute__((ext_vector_type(16))) __bf16 v16bf;
typedef __attribute__((ext_vector_type(8)))  __bf16 v8bf;
typedef __attribute__((ext_vector_type(8)))  float  v8f;

#define B_  16
#define SD_ 768
#define TD_ 1024
#define HW_ 32
#define N_  1024            // tokens per image
#define HEADS_ 8
#define HD_ 128
#define FFN_ 4096
#define KCONV_ (SD_*9)      // 6912

// ---------------------------------------------------------------------------
// Generic WMMA GEMM: D = epilogue(A @ B)
//   block tile 128(M) x 128(N), K-step 32, 256 threads = 8 waves (4 M x 2 N),
//   each wave owns 32x64: 2 A frags x 4 B frags = 8 WMMAs per K-step.
//   Double-buffered LDS; global loads register-staged ahead of compute;
//   global_prefetch for the K+64 tile.
// ---------------------------------------------------------------------------
struct GemmP {
    const bf16* A; const bf16* Bm; void* D;
    int M, N, K, lda, ldb, ldd;
    int zH, zW;                                   // z = ((b*zH)+h)*zW + w
    long aB, aH, aW, bB, bH, bW, dB, dH, dW;      // per-axis element offsets
    const float* bias; const float* scale; const float* shift;
    const float* pos;  const float* resid;
    const int* rowmap;                            // optional output row remap
    float alpha;
};

static __device__ inline v16bf frag_ld(const bf16* base) {
    const v8bf* p = (const v8bf*)base;            // 16B aligned by construction
    v8bf lo = p[0];                               // k = half*8 + 0..7
    v8bf hi = p[2];                               // k = 16 + half*8 + 0..7
    v16bf r;
#pragma unroll
    for (int i = 0; i < 8; ++i) { r[i] = lo[i]; r[i + 8] = hi[i]; }
    return r;
}

// AMODE: 0 = plain row-major A, 1 = implicit im2col conv3x3 from padded NHWC
// BTRANS: 0 = B is [K][N], 1 = B is [N][K]
// EPI: 0 = *alpha, 1 = +bias, 2 = +bias,ReLU, 3 = conv BN+ReLU+pos, 4 = +bias+resid
// OUTBF: 1 = store bf16, 0 = store f32
template <int AMODE, int BTRANS, int EPI, int OUTBF>
__global__ __launch_bounds__(256) void gemm_wmma(GemmP p) {
    __shared__ __align__(16) bf16 sA[2][128 * 32];
    __shared__ __align__(16) bf16 sBt[2][128 * 32];   // transposed: [n*32 + k]

    const int z  = blockIdx.z;
    const int zb = z / (p.zH * p.zW);
    const int zr = z % (p.zH * p.zW);
    const int zh = zr / p.zW;
    const int zw = zr % p.zW;
    const long offA = (long)zb * p.aB + (long)zh * p.aH + (long)zw * p.aW;
    const long offB = (long)zb * p.bB + (long)zh * p.bH + (long)zw * p.bW;
    const long offD = (long)zb * p.dB + (long)zh * p.dH + (long)zw * p.dW;

    const int bm = blockIdx.x * 128;
    const int bn = blockIdx.y * 128;
    const int t  = threadIdx.x;
    const int lane = t & 31, wid = t >> 5;
    const int hl = lane >> 4, idx = lane & 15;
    const int mw = wid & 3;        // M slab (32 rows)
    const int nw = wid >> 2;       // N half (64 cols)

    // staging geometry
    const int am  = t >> 1, akc = (t & 1) * 16;      // A: 128 rows x 2 chunks
    const int bk0 = t >> 3, bnc = (t & 7) * 16;      // B (KN): 32 k x 8 n-chunks

    auto aAddr = [&](int kt) -> const bf16* {
        if (AMODE == 0) {
            return p.A + offA + (size_t)(bm + am) * p.lda + kt + akc;
        } else {
            const int mg = bm + am;
            const int bi = mg >> 10, py = (mg >> 5) & 31, px = mg & 31;
            const int kg = kt + akc;
            const int s  = kg / SD_, ci = kg - s * SD_;
            const int dy = s / 3,  dx = s - dy * 3;
            return p.A + ((size_t)((bi * 34 + py + dy) * 34 + (px + dx))) * SD_ + ci;
        }
    };
    auto bAddr = [&](int kt) -> const bf16* {
        if (BTRANS) return p.Bm + offB + (size_t)(bn + (t >> 1)) * p.ldb + kt + (t & 1) * 16;
        else        return p.Bm + offB + (size_t)(kt + bk0) * p.ldb + bn + bnc;
    };

    v8bf ra[2], rb[2];
    auto stage_load = [&](int kt) {
        const v8bf* av = (const v8bf*)aAddr(kt);
        ra[0] = av[0]; ra[1] = av[1];
        const v8bf* bv = (const v8bf*)bAddr(kt);
        rb[0] = bv[0]; rb[1] = bv[1];
    };
    auto stage_store = [&](int buf) {
        v8bf* dv = (v8bf*)(&sA[buf][am * 32 + akc]);
        dv[0] = ra[0]; dv[1] = ra[1];
        if (BTRANS) {
            v8bf* bv = (v8bf*)(&sBt[buf][(t >> 1) * 32 + (t & 1) * 16]);
            bv[0] = rb[0]; bv[1] = rb[1];
        } else {
#pragma unroll
            for (int j = 0; j < 8; ++j) {
                sBt[buf][(bnc + j)     * 32 + bk0] = rb[0][j];
                sBt[buf][(bnc + 8 + j) * 32 + bk0] = rb[1][j];
            }
        }
    };

    v8f acc[2][4] = {};

    // prologue: stage K-tile 0
    stage_load(0);
    stage_store(0);
    __syncthreads();

    int cur = 0;
    for (int kt = 0; kt < p.K; kt += 32) {
        const bool nxt = (kt + 32) < p.K;
        if (nxt) {
            stage_load(kt + 32);                  // global loads, ahead of math
            if (kt + 64 < p.K) {                  // pull K+64 toward cache
                __builtin_prefetch(aAddr(kt + 64), 0, 3);
                __builtin_prefetch(bAddr(kt + 64), 0, 3);
            }
        }
        // ---- 8 WMMAs from buffer `cur` ----
        v16bf a0 = frag_ld(&sA[cur][(mw * 32      + idx) * 32 + hl * 8]);
        v16bf a1 = frag_ld(&sA[cur][(mw * 32 + 16 + idx) * 32 + hl * 8]);
#pragma unroll
        for (int j = 0; j < 4; ++j) {
            v16bf b = frag_ld(&sBt[cur][(nw * 64 + j * 16 + idx) * 32 + hl * 8]);
            acc[0][j] = __builtin_amdgcn_wmma_f32_16x16x32_bf16(
                false, a0, false, b, (short)0, acc[0][j], false, false);
            acc[1][j] = __builtin_amdgcn_wmma_f32_16x16x32_bf16(
                false, a1, false, b, (short)0, acc[1][j], false, false);
        }
        if (nxt) stage_store(cur ^ 1);
        __syncthreads();
        cur ^= 1;
    }

    // ---- epilogue ----
#pragma unroll
    for (int i = 0; i < 2; ++i) {
        const int row0 = bm + mw * 32 + i * 16 + hl * 8;
#pragma unroll
        for (int j = 0; j < 4; ++j) {
            const int col = bn + nw * 64 + j * 16 + idx;
            float badd = 0.f, sc = 1.f, sh = 0.f;
            if (EPI == 1 || EPI == 2 || EPI == 4) badd = p.bias[col];
            if (EPI == 3) { sc = p.scale[col]; sh = p.shift[col]; }
#pragma unroll
            for (int r = 0; r < 8; ++r) {
                const int row = row0 + r;
                float v = acc[i][j][r] * p.alpha;
                if (EPI == 1 || EPI == 2 || EPI == 4) v += badd;
                if (EPI == 2) v = v > 0.f ? v : 0.f;
                if (EPI == 3) {
                    v = v * sc + sh;
                    v = v > 0.f ? v : 0.f;
                    v += p.pos[(size_t)col * N_ + (row & (N_ - 1))];
                }
                if (EPI == 4) v += p.resid[(size_t)row * p.ldd + col];
                const long orow = p.rowmap ? (long)p.rowmap[row] : (long)row;
                const size_t di = (size_t)offD + (size_t)orow * p.ldd + col;
                if (OUTBF) ((bf16*)p.D)[di] = (bf16)v;
                else       ((float*)p.D)[di] = v;
            }
        }
    }
}

// ---------------------------------------------------------------------------
// Helper / fusion kernels
// ---------------------------------------------------------------------------
__global__ __launch_bounds__(256) void pad_nchw_to_nhwc_bf16(
        const float* __restrict__ x, bf16* __restrict__ xp, int total) {
    int i = blockIdx.x * 256 + threadIdx.x;
    if (i >= total) return;
    int c = i % SD_; int tpx = i / SD_;
    int px = tpx % 34; int tpy = tpx / 34;
    int py = tpy % 34; int b  = tpy / 34;
    float v = 0.f;
    if (py >= 1 && py <= 32 && px >= 1 && px <= 32)
        v = x[((size_t)(b * SD_ + c) * HW_ + (py - 1)) * HW_ + (px - 1)];
    xp[i] = (bf16)v;
}

__global__ __launch_bounds__(256) void conv_w_transform(
        const float* __restrict__ w, bf16* __restrict__ wc, int total) {
    int i = blockIdx.x * 256 + threadIdx.x;
    if (i >= total) return;
    int o = i % SD_; int k = i / SD_;       // k = (dy*3+dx)*768 + ci
    int s = k / SD_; int ci = k - s * SD_;
    int dy = s / 3;  int dx = s - dy * 3;
    wc[i] = (bf16)w[((size_t)(o * SD_ + ci) * 3 + dy) * 3 + dx];
}

__global__ void bn_fold(const float* g, const float* be, const float* mean,
                        const float* var, const float* bias,
                        float* scale, float* shift) {
    int c = blockIdx.x * 256 + threadIdx.x;
    if (c >= SD_) return;
    float s = g[c] * rsqrtf(var[c] + 1e-5f);
    scale[c] = s;
    shift[c] = be[c] + (bias[c] - mean[c]) * s;
}

__global__ __launch_bounds__(256) void f32_to_bf16_k(
        const float* __restrict__ in, bf16* __restrict__ out, int n) {
    int i = blockIdx.x * 256 + threadIdx.x;
    if (i < n) out[i] = (bf16)in[i];
}

__global__ void build_rowmap(int* __restrict__ map, int total) {
    int m = blockIdx.x * 256 + threadIdx.x;
    if (m >= total) return;
    int n = m & (N_ - 1);
    int y = n >> 5, x = n & 31;
    int ay = y >> 4, iy = y & 15, ax = x >> 4, ix = x & 15;
    map[m] = (m & ~(N_ - 1)) + ((ay * 2 + ax) << 8) + (iy << 4) + ix;
}

__global__ __launch_bounds__(256) void softmax_rows(bf16* __restrict__ S) {
    __shared__ float red[256];
    const size_t row = blockIdx.x;
    bf16* p = S + row * 256;
    const int t = threadIdx.x;
    float v = (float)p[t];
    red[t] = v; __syncthreads();
    for (int s = 128; s > 0; s >>= 1) {
        if (t < s) red[t] = fmaxf(red[t], red[t + s]);
        __syncthreads();
    }
    const float mx = red[0]; __syncthreads();
    const float e = __expf(v - mx);
    red[t] = e; __syncthreads();
    for (int s = 128; s > 0; s >>= 1) {
        if (t < s) red[t] += red[t + s];
        __syncthreads();
    }
    p[t] = (bf16)(e / red[0]);
}

__global__ __launch_bounds__(256) void layernorm_rows(
        float* __restrict__ fea, bf16* __restrict__ hbf,
        const float* __restrict__ g, const float* __restrict__ b) {
    __shared__ float r1[256], r2[256];
    const size_t row = blockIdx.x;
    float* p = fea + row * TD_;
    const int t = threadIdx.x;
    float x0 = p[t], x1 = p[t + 256], x2 = p[t + 512], x3 = p[t + 768];
    r1[t] = x0 + x1 + x2 + x3;
    r2[t] = x0 * x0 + x1 * x1 + x2 * x2 + x3 * x3;
    __syncthreads();
    for (int s = 128; s > 0; s >>= 1) {
        if (t < s) { r1[t] += r1[t + s]; r2[t] += r2[t + s]; }
        __syncthreads();
    }
    const float mu = r1[0] * (1.f / TD_);
    const float var = r2[0] * (1.f / TD_) - mu * mu;
    const float rs = rsqrtf(var + 1e-5f);
    float xs[4] = {x0, x1, x2, x3};
#pragma unroll
    for (int i = 0; i < 4; ++i) {
        const int c = t + i * 256;
        const float y = (xs[i] - mu) * rs * g[c] + b[c];
        p[c] = y;                                // residual (f32, in-place)
        hbf[row * TD_ + c] = (bf16)y;            // FFN input (bf16)
    }
}

// ---------------------------------------------------------------------------
// Host launcher
// ---------------------------------------------------------------------------
static void launch_gemm_common(GemmP& p) {
    p.zH = p.zH ? p.zH : 1; p.zW = p.zW ? p.zW : 1;
}

extern "C" void kernel_launch(void* const* d_in, const int* in_sizes, int n_in,
                              void* d_out, int out_size, void* d_ws, size_t ws_size,
                              hipStream_t stream) {
    const float* x     = (const float*)d_in[0];
    const float* csw   = (const float*)d_in[1];
    const float* csb   = (const float*)d_in[2];
    const float* bng   = (const float*)d_in[3];
    const float* bnb   = (const float*)d_in[4];
    const float* bnm   = (const float*)d_in[5];
    const float* bnv   = (const float*)d_in[6];
    const float* pos   = (const float*)d_in[7];
    const float* qwght = (const float*)d_in[8];
    const float* cpw   = (const float*)d_in[9];
    const float* cpb   = (const float*)d_in[10];
    const float* wq    = (const float*)d_in[11];
    const float* bq    = (const float*)d_in[12];
    const float* wk    = (const float*)d_in[13];
    const float* bk    = (const float*)d_in[14];
    const float* wvp   = (const float*)d_in[15];
    const float* bv    = (const float*)d_in[16];
    const float* wpr   = (const float*)d_in[17];
    const float* bpr   = (const float*)d_in[18];
    const float* lng   = (const float*)d_in[19];
    const float* lnb   = (const float*)d_in[20];
    const float* w1    = (const float*)d_in[21];
    const float* b1    = (const float*)d_in[22];
    const float* w2    = (const float*)d_in[23];
    const float* b2    = (const float*)d_in[24];
    float* out = (float*)d_out;
    (void)in_sizes; (void)n_in; (void)out_size; (void)ws_size;

    // ---- workspace bump allocator (~500 MB) ----
    char* wsb = (char*)d_ws; size_t off = 0;
    auto alloc = [&](size_t bytes) -> void* {
        void* p = wsb + off;
        off = (off + bytes + 255) & ~(size_t)255;
        return p;
    };
    const int M_ALL = B_ * N_;                               // 16384
    bf16*  xp      = (bf16*) alloc((size_t)B_ * 34 * 34 * SD_ * 2);
    bf16*  wconv   = (bf16*) alloc((size_t)KCONV_ * SD_ * 2);
    float* bnscale = (float*)alloc(SD_ * 4);
    float* bnshift = (float*)alloc(SD_ * 4);
    bf16*  x_tok   = (bf16*) alloc((size_t)M_ALL * SD_ * 2);
    bf16*  qw_bf   = (bf16*) alloc((size_t)N_ * TD_ * 2);
    bf16*  wp_bf   = (bf16*) alloc((size_t)TD_ * TD_ * 2);
    bf16*  pos_emb = (bf16*) alloc((size_t)N_ * TD_ * 2);
    bf16*  wq_bf   = (bf16*) alloc((size_t)TD_ * TD_ * 2);
    bf16*  wk_bf   = (bf16*) alloc((size_t)SD_ * TD_ * 2);
    bf16*  wv_bf   = (bf16*) alloc((size_t)SD_ * TD_ * 2);
    bf16*  wpr_bf  = (bf16*) alloc((size_t)TD_ * TD_ * 2);
    bf16*  w1_bf   = (bf16*) alloc((size_t)TD_ * FFN_ * 2);
    bf16*  w2_bf   = (bf16*) alloc((size_t)FFN_ * TD_ * 2);
    bf16*  qmat    = (bf16*) alloc((size_t)N_ * TD_ * 2);
    bf16*  kmat    = (bf16*) alloc((size_t)M_ALL * TD_ * 2);
    bf16*  vmat    = (bf16*) alloc((size_t)M_ALL * TD_ * 2);
    bf16*  Smat    = (bf16*) alloc((size_t)512 * 256 * 256 * 2);
    bf16*  omat    = (bf16*) alloc((size_t)M_ALL * TD_ * 2);
    float* fea     = (float*)alloc((size_t)M_ALL * TD_ * 4);  // becomes h (f32)
    bf16*  h_bf    = (bf16*) alloc((size_t)M_ALL * TD_ * 2);
    bf16*  hid     = (bf16*) alloc((size_t)M_ALL * FFN_ * 2);
    int*   rowmap  = (int*)  alloc((size_t)M_ALL * 4);

    auto ew = [&](int n) { return dim3((n + 255) / 256); };

    // ---- prep ----
    {
        int n = B_ * 34 * 34 * SD_;
        pad_nchw_to_nhwc_bf16<<<ew(n), 256, 0, stream>>>(x, xp, n);
    }
    {
        int n = KCONV_ * SD_;
        conv_w_transform<<<ew(n), 256, 0, stream>>>(csw, wconv, n);
    }
    bn_fold<<<ew(SD_), 256, 0, stream>>>(bng, bnb, bnm, bnv, csb, bnscale, bnshift);
    f32_to_bf16_k<<<ew(N_ * TD_), 256, 0, stream>>>(qwght, qw_bf, N_ * TD_);
    f32_to_bf16_k<<<ew(TD_ * TD_), 256, 0, stream>>>(cpw, wp_bf, TD_ * TD_);
    f32_to_bf16_k<<<ew(TD_ * TD_), 256, 0, stream>>>(wq, wq_bf, TD_ * TD_);
    f32_to_bf16_k<<<ew(SD_ * TD_), 256, 0, stream>>>(wk, wk_bf, SD_ * TD_);
    f32_to_bf16_k<<<ew(SD_ * TD_), 256, 0, stream>>>(wvp, wv_bf, SD_ * TD_);
    f32_to_bf16_k<<<ew(TD_ * TD_), 256, 0, stream>>>(wpr, wpr_bf, TD_ * TD_);
    f32_to_bf16_k<<<ew(TD_ * FFN_), 256, 0, stream>>>(w1, w1_bf, TD_ * FFN_);
    f32_to_bf16_k<<<ew(FFN_ * TD_), 256, 0, stream>>>(w2, w2_bf, FFN_ * TD_);
    build_rowmap<<<ew(M_ALL), 256, 0, stream>>>(rowmap, M_ALL);

    GemmP p{};

    // ---- 1) conv3x3 + BN + ReLU + pos  ->  x_tok bf16 [16384][768] ----
    p = GemmP{};
    p.A = xp; p.Bm = wconv; p.D = x_tok;
    p.M = M_ALL; p.N = SD_; p.K = KCONV_;
    p.lda = KCONV_; p.ldb = SD_; p.ldd = SD_;
    p.scale = bnscale; p.shift = bnshift; p.pos = pos; p.alpha = 1.f;
    launch_gemm_common(p);
    gemm_wmma<1, 0, 3, 1><<<dim3(M_ALL / 128, SD_ / 128, 1), 256, 0, stream>>>(p);

    // ---- 2) pos path conv1x1 + ReLU (batch-shared) -> pos_emb [1024][1024] ----
    p = GemmP{};
    p.A = qw_bf; p.Bm = wp_bf; p.D = pos_emb;
    p.M = N_; p.N = TD_; p.K = TD_;
    p.lda = TD_; p.ldb = TD_; p.ldd = TD_;
    p.bias = cpb; p.alpha = 1.f;
    launch_gemm_common(p);
    gemm_wmma<0, 1, 2, 1><<<dim3(N_ / 128, TD_ / 128, 1), 256, 0, stream>>>(p);

    // ---- 3) q projection (batch-shared), rows stored window-major ----
    p = GemmP{};
    p.A = pos_emb; p.Bm = wq_bf; p.D = qmat;
    p.M = N_; p.N = TD_; p.K = TD_;
    p.lda = TD_; p.ldb = TD_; p.ldd = TD_;
    p.bias = bq; p.rowmap = rowmap; p.alpha = 1.f;
    launch_gemm_common(p);
    gemm_wmma<0, 0, 1, 1><<<dim3(N_ / 128, TD_ / 128, 1), 256, 0, stream>>>(p);

    // ---- 4) k projection, rows window-major per image ----
    p = GemmP{};
    p.A = x_tok; p.Bm = wk_bf; p.D = kmat;
    p.M = M_ALL; p.N = TD_; p.K = SD_;
    p.lda = SD_; p.ldb = TD_; p.ldd = TD_;
    p.bias = bk; p.rowmap = rowmap; p.alpha = 1.f;
    launch_gemm_common(p);
    gemm_wmma<0, 0, 1, 1><<<dim3(M_ALL / 128, TD_ / 128, 1), 256, 0, stream>>>(p);

    // ---- 5) v projection ----
    p.Bm = wv_bf; p.bias = bv; p.D = vmat;
    gemm_wmma<0, 0, 1, 1><<<dim3(M_ALL / 128, TD_ / 128, 1), 256, 0, stream>>>(p);

    // ---- 6) S = alpha * q @ k^T   batched over z=(b,h,w), 256x256x128 ----
    p = GemmP{};
    p.A = qmat; p.Bm = kmat; p.D = Smat;
    p.M = 256; p.N = 256; p.K = HD_;
    p.lda = TD_; p.ldb = TD_; p.ldd = 256;
    p.zH = HEADS_; p.zW = 4;
    p.aB = 0;                 p.aH = HD_;            p.aW = 256L * TD_;
    p.bB = (long)N_ * TD_;    p.bH = HD_;            p.bW = 256L * TD_;
    p.dB = 32L * 65536;       p.dH = 4L * 65536;     p.dW = 65536;
    p.alpha = 5.0f / sqrtf((float)HD_);
    gemm_wmma<0, 1, 0, 1><<<dim3(2, 2, 512), 256, 0, stream>>>(p);

    // ---- 7) softmax rows (in place over Smat) ----
    softmax_rows<<<dim3(512 * 256), 256, 0, stream>>>(Smat);

    // ---- 8) O = P @ v   batched, 256x128x256 -> omat [16384][1024] ----
    p = GemmP{};
    p.A = Smat; p.Bm = vmat; p.D = omat;
    p.M = 256; p.N = HD_; p.K = 256;
    p.lda = 256; p.ldb = TD_; p.ldd = TD_;
    p.zH = HEADS_; p.zW = 4;
    p.aB = 32L * 65536;       p.aH = 4L * 65536;     p.aW = 65536;
    p.bB = (long)N_ * TD_;    p.bH = HD_;            p.bW = 256L * TD_;
    p.dB = (long)N_ * TD_;    p.dH = HD_;            p.dW = 256L * TD_;
    p.alpha = 1.f;
    gemm_wmma<0, 0, 0, 1><<<dim3(2, 1, 512), 256, 0, stream>>>(p);

    // ---- 9) output projection -> fea f32 [16384][1024] ----
    p = GemmP{};
    p.A = omat; p.Bm = wpr_bf; p.D = fea;
    p.M = M_ALL; p.N = TD_; p.K = TD_;
    p.lda = TD_; p.ldb = TD_; p.ldd = TD_;
    p.bias = bpr; p.alpha = 1.f;
    launch_gemm_common(p);
    gemm_wmma<0, 0, 1, 0><<<dim3(M_ALL / 128, TD_ / 128, 1), 256, 0, stream>>>(p);

    // ---- 10) LayerNorm -> h (f32, in place) + h_bf (bf16) ----
    layernorm_rows<<<dim3(M_ALL), 256, 0, stream>>>(fea, h_bf, lng, lnb);

    // ---- 11) FFN1: relu(h @ w1 + b1) -> hid bf16 [16384][4096] ----
    p = GemmP{};
    p.A = h_bf; p.Bm = w1_bf; p.D = hid;
    p.M = M_ALL; p.N = FFN_; p.K = TD_;
    p.lda = TD_; p.ldb = FFN_; p.ldd = FFN_;
    p.bias = b1; p.alpha = 1.f;
    launch_gemm_common(p);
    gemm_wmma<0, 0, 2, 1><<<dim3(M_ALL / 128, FFN_ / 128, 1), 256, 0, stream>>>(p);

    // ---- 12) FFN2 + residual h -> d_out f32 [16384][1024] ----
    p = GemmP{};
    p.A = hid; p.Bm = w2_bf; p.D = out;
    p.M = M_ALL; p.N = TD_; p.K = FFN_;
    p.lda = FFN_; p.ldb = TD_; p.ldd = TD_;
    p.bias = b2; p.resid = fea; p.alpha = 1.f;
    launch_gemm_common(p);
    gemm_wmma<0, 0, 4, 0><<<dim3(M_ALL / 128, TD_ / 128, 1), 256, 0, stream>>>(p);
}
